// RWKV_TimeMix_51754355916913
// MI455X (gfx1250) — matrix operand order repaired and checked by reading, not compiled, not run on hardware
//
#include <hip/hip_runtime.h>
#include <hip/hip_bf16.h>

// ---------------------------------------------------------------------------
// RWKV-4 TimeMix, MI455X (gfx1250, wave32).
//   1) weight_swizzle  : Wk/Wv/Wr/Wo fp32 -> bf16 in WMMA B-fragment order.
//   2) time_mix_prep   : time-shift + mix, fp32 -> bf16 xk/xv/xr.
//   3) gemm_bf16_wmma  : V_WMMA_F32_16X16X32_BF16, A staged via
//                        GLOBAL_LOAD_ASYNC_TO_LDS_B128 double-buffered pipeline.
//   4) wkv_pass1/2/3   : chunked (S=16) numerically-stable WKV scan.
//   5) rwkv_prep       : sigmoid(r)*wkv -> bf16.
//   6) gemm_bf16_wmma  : output projection -> d_out (fp32).
// ---------------------------------------------------------------------------

typedef __bf16 bf16x16 __attribute__((ext_vector_type(16)));
typedef __bf16 bf16x8  __attribute__((ext_vector_type(8)));
typedef float  f32x8   __attribute__((ext_vector_type(8)));

__device__ __forceinline__ unsigned short f2bfu(float f) {
    unsigned u = __float_as_uint(f);
    unsigned r = u + 0x7FFFu + ((u >> 16) & 1u);   // round-to-nearest-even
    return (unsigned short)(r >> 16);
}

// ---------------------------------------------------------------------------
// 1) Weight pre-swizzle: W [N,K] fp32 row-major -> bf16 in B-fragment order.
//   element (lane l, dword j, half h) of tile (nb,kb) = W[n][k] with
//   n = nb*16 + (l & 15),  k = kb*32 + (l>>4)*16 + 2*j + h
// ---------------------------------------------------------------------------
__global__ void weight_swizzle(const float* __restrict__ W,
                               unsigned short* __restrict__ out,
                               int K, int total_dwords) {
    int d = blockIdx.x * blockDim.x + threadIdx.x;
    if (d >= total_dwords) return;
    int tile = d >> 8;          // 256 dwords per tile
    int wd   = d & 255;
    int l    = wd >> 3;         // lane 0..31
    int j    = wd & 7;          // dword 0..7
    int ktiles = K >> 5;
    int nb = tile / ktiles;
    int kb = tile % ktiles;
    int n  = nb * 16 + (l & 15);
    int k  = kb * 32 + ((l >> 4) * 16) + 2 * j;
    float a = W[(size_t)n * K + k];
    float b = W[(size_t)n * K + k + 1];
    out[2 * (size_t)d]     = f2bfu(a);
    out[2 * (size_t)d + 1] = f2bfu(b);
}

// ---------------------------------------------------------------------------
// 2) Time-shift + mix -> bf16 (float4 / ushort4 vectorized)
// ---------------------------------------------------------------------------
__global__ void time_mix_prep(const float* __restrict__ x,
                              const float* __restrict__ tmk,
                              const float* __restrict__ tmv,
                              const float* __restrict__ tmr,
                              unsigned short* __restrict__ xk,
                              unsigned short* __restrict__ xv,
                              unsigned short* __restrict__ xr,
                              int T, int C, size_t n4) {
    size_t i = (size_t)blockIdx.x * blockDim.x + threadIdx.x;
    if (i >= n4) return;
    size_t idx = i * 4;
    int c = (int)(idx % C);
    int t = (int)((idx / C) % T);

    float4 xc = *(const float4*)(x + idx);
    float4 xp = make_float4(0.f, 0.f, 0.f, 0.f);
    if (t > 0) xp = *(const float4*)(x + idx - C);

    float4 mk = *(const float4*)(tmk + c);
    float4 mv = *(const float4*)(tmv + c);
    float4 mr = *(const float4*)(tmr + c);

    ushort4 ok, ov, orr;
    ok.x = f2bfu(xc.x * mk.x + xp.x * (1.f - mk.x));
    ok.y = f2bfu(xc.y * mk.y + xp.y * (1.f - mk.y));
    ok.z = f2bfu(xc.z * mk.z + xp.z * (1.f - mk.z));
    ok.w = f2bfu(xc.w * mk.w + xp.w * (1.f - mk.w));
    ov.x = f2bfu(xc.x * mv.x + xp.x * (1.f - mv.x));
    ov.y = f2bfu(xc.y * mv.y + xp.y * (1.f - mv.y));
    ov.z = f2bfu(xc.z * mv.z + xp.z * (1.f - mv.z));
    ov.w = f2bfu(xc.w * mv.w + xp.w * (1.f - mv.w));
    orr.x = f2bfu(xc.x * mr.x + xp.x * (1.f - mr.x));
    orr.y = f2bfu(xc.y * mr.y + xp.y * (1.f - mr.y));
    orr.z = f2bfu(xc.z * mr.z + xp.z * (1.f - mr.z));
    orr.w = f2bfu(xc.w * mr.w + xp.w * (1.f - mr.w));
    *(ushort4*)(xk + idx) = ok;
    *(ushort4*)(xv + idx) = ov;
    *(ushort4*)(xr + idx) = orr;
}

// ---------------------------------------------------------------------------
// 3) bf16 WMMA GEMM, fixed 16384 x 1024 x 1024.
//    Block = 128x128, 8 waves (4 M x 2 N), each wave 2x4 WMMA tiles, K-step 32.
//    A staged into double-buffered LDS via GLOBAL_LOAD_ASYNC_TO_LDS_B128
//    (ASYNCcnt pipeline, one barrier per K-step). B fragments are contiguous
//    32B-per-lane reads of the pre-swizzled (L2-resident) weights.
// ---------------------------------------------------------------------------
#define MBLK 128
#define NBLK 128
#define A_LDS_STRIDE 40   // 32 + 8 bf16 padding -> conflict-free fragment reads

constexpr int GEMM_M  = 16384;
constexpr int GEMM_N  = 1024;
constexpr int GEMM_K  = 1024;
constexpr int KTILES  = GEMM_K / 32;                       // 32
constexpr unsigned A_BUF_BYTES = MBLK * A_LDS_STRIDE * 2;  // 10240 per buffer

__global__ __launch_bounds__(256) void gemm_bf16_wmma(
    const unsigned short* __restrict__ A,     // [M,K] bf16
    const unsigned short* __restrict__ Bsw,   // swizzled weights
    float* __restrict__ Cc) {                 // [M,N] f32
    __shared__ alignas(16) unsigned short As[2 * MBLK * A_LDS_STRIDE];

    const int tid   = threadIdx.x;
    const int lane  = tid & 31;
    const int wave  = tid >> 5;
    const int waveM = wave & 3;    // 0..3
    const int waveN = wave >> 2;   // 0..1
    const int m0 = blockIdx.y * MBLK;
    const int n0 = blockIdx.x * NBLK;
    const int lm = lane & 15;
    const int lh = lane >> 4;

    // per-thread async-copy slot: 32 bytes of the 128x32 A tile
    const int r   = tid >> 1;      // row 0..127
    const int seg = tid & 1;       // 16-bf16 segment
    // LDS-aperture flat address carries the LDS byte offset in bits [31:0]
    const unsigned ldsBase = (unsigned)(uintptr_t)(&As[0]);
    const unsigned myLds   = ldsBase + (unsigned)(r * A_LDS_STRIDE + seg * 16) * 2u;
    const unsigned short* gsrc = A + (size_t)(m0 + r) * GEMM_K + seg * 16;

    f32x8 acc[2][4];
#pragma unroll
    for (int i = 0; i < 2; ++i)
#pragma unroll
        for (int j = 0; j < 4; ++j)
#pragma unroll
            for (int e = 0; e < 8; ++e) acc[i][j][e] = 0.f;

    // prefetch K-tile 0 into buffer 0 (offset field applies to LDS and memory)
    {
        unsigned long long ga = (unsigned long long)(uintptr_t)gsrc;
        asm volatile("global_load_async_to_lds_b128 %0, %1, off"
                     :: "v"(myLds), "v"(ga) : "memory");
        asm volatile("global_load_async_to_lds_b128 %0, %1, off offset:16"
                     :: "v"(myLds), "v"(ga) : "memory");
    }

    for (int kb = 0; kb < KTILES; ++kb) {
        // own async writes for the current tile done; barrier also retires all
        // waves' LDS reads of the *other* buffer (DScnt waited) before refill
        asm volatile("s_wait_asynccnt 0x0" ::: "memory");
        __syncthreads();

        if (kb + 1 < KTILES) {
            unsigned long long ga =
                (unsigned long long)(uintptr_t)(gsrc + (kb + 1) * 32);
            unsigned dst = myLds + ((unsigned)(kb + 1) & 1u) * A_BUF_BYTES;
            asm volatile("global_load_async_to_lds_b128 %0, %1, off"
                         :: "v"(dst), "v"(ga) : "memory");
            asm volatile("global_load_async_to_lds_b128 %0, %1, off offset:16"
                         :: "v"(dst), "v"(ga) : "memory");
        }

        const unsigned short* abase = &As[(kb & 1) * MBLK * A_LDS_STRIDE];

        // A fragments (contiguous k order to match swizzled B: lanes 0-15 K0-15,
        // lanes 16-31 K16-31)
        bf16x16 afrag[2];
#pragma unroll
        for (int i = 0; i < 2; ++i) {
            int row = waveM * 32 + i * 16 + lm;
            const unsigned short* p = abase + row * A_LDS_STRIDE + lh * 16;
            bf16x8 lo = *(const bf16x8*)(p);
            bf16x8 hi = *(const bf16x8*)(p + 8);
            afrag[i] = __builtin_shufflevector(lo, hi, 0, 1, 2, 3, 4, 5, 6, 7,
                                               8, 9, 10, 11, 12, 13, 14, 15);
        }

        // B fragments: contiguous 32B per lane from swizzled weights
        bf16x16 bfrag[4];
#pragma unroll
        for (int j = 0; j < 4; ++j) {
            int nb = (n0 >> 4) + waveN * 4 + j;
            const unsigned short* p =
                Bsw + ((size_t)nb * KTILES + kb) * 512 + lane * 16;
            bfrag[j] = *(const bf16x16*)p;
        }

#pragma unroll
        for (int i = 0; i < 2; ++i)
#pragma unroll
            for (int j = 0; j < 4; ++j)
                acc[i][j] = __builtin_amdgcn_wmma_f32_16x16x32_bf16(
                    false, afrag[i], false, bfrag[j],
                    (short)0, acc[i][j], false, false);
    }

    // epilogue: VGPR g holds M=g (lanes 0-15) / M=8+g (lanes 16-31);
    // compile-time GEMM_N lets g*GEMM_N fold into store immediate offsets
#pragma unroll
    for (int i = 0; i < 2; ++i)
#pragma unroll
        for (int j = 0; j < 4; ++j) {
            float* p = Cc + (size_t)(m0 + waveM * 32 + i * 16 + 8 * lh) * GEMM_N
                          + (n0 + waveN * 64 + j * 16 + lm);
#pragma unroll
            for (int g = 0; g < 8; ++g)
                p[(size_t)g * GEMM_N] = acc[i][j][g];
        }
}

// ---------------------------------------------------------------------------
// 4) Chunked WKV scan. State (aa,bb,pp) means contribution (aa,bb)*exp(pp).
// ---------------------------------------------------------------------------
__global__ void wkv_pass1(const float* __restrict__ k, const float* __restrict__ v,
                          const float* __restrict__ tdec,
                          float* __restrict__ stA, float* __restrict__ stB,
                          float* __restrict__ stP,
                          int Bn, int T, int C, int S) {
    int i = blockIdx.x * blockDim.x + threadIdx.x;   // (s*Bn + b)*C + c
    if (i >= S * Bn * C) return;
    int c = i % C;
    int s = i / (Bn * C);
    int b = (i / C) % Bn;
    float w = -__expf(tdec[c]);
    int L = T / S;
    const size_t base = ((size_t)b * T + (size_t)s * L) * C + c;
    float aa = 0.f, bb = 0.f, pp = -1e38f;
    for (int t = 0; t < L; ++t) {
        float kt = k[base + (size_t)t * C];
        float vt = v[base + (size_t)t * C];
        float ww = pp + w;
        float q  = fmaxf(ww, kt);
        float e1 = __expf(ww - q);
        float e2 = __expf(kt - q);
        aa = e1 * aa + e2 * vt;
        bb = e1 * bb + e2;
        pp = q;
    }
    stA[i] = aa; stB[i] = bb; stP[i] = pp;
}

__global__ void wkv_pass2(const float* __restrict__ stA, const float* __restrict__ stB,
                          const float* __restrict__ stP,
                          float* __restrict__ inA, float* __restrict__ inB,
                          float* __restrict__ inP,
                          const float* __restrict__ tdec,
                          int Bn, int T, int C, int S) {
    int i = blockIdx.x * blockDim.x + threadIdx.x;   // b*C + c
    if (i >= Bn * C) return;
    int c = i % C;
    float w  = -__expf(tdec[c]);
    float Lw = (float)(T / S) * w;
    float a = 0.f, b = 0.f, p = -1e38f;
    for (int s = 0; s < S; ++s) {
        size_t idx = (size_t)s * Bn * C + i;
        inA[idx] = a; inB[idx] = b; inP[idx] = p;
        float ca = stA[idx], cb = stB[idx], cp = stP[idx];
        float ps = p + Lw;
        float q  = fmaxf(ps, cp);
        float e1 = __expf(ps - q);
        float e2 = __expf(cp - q);
        a = e1 * a + e2 * ca;
        b = e1 * b + e2 * cb;
        p = q;
    }
}

__global__ void wkv_pass3(const float* __restrict__ k, const float* __restrict__ v,
                          const float* __restrict__ inA, const float* __restrict__ inB,
                          const float* __restrict__ inP,
                          const float* __restrict__ tdec, const float* __restrict__ tfir,
                          float* __restrict__ wkv,
                          int Bn, int T, int C, int S) {
    int i = blockIdx.x * blockDim.x + threadIdx.x;   // (s*Bn + b)*C + c
    if (i >= S * Bn * C) return;
    int c = i % C;
    int s = i / (Bn * C);
    int b = (i / C) % Bn;
    float w = -__expf(tdec[c]);
    float u = tfir[c];
    int L = T / S;
    float aa = inA[i], bb = inB[i], pp = inP[i];
    const size_t base = ((size_t)b * T + (size_t)s * L) * C + c;
    for (int t = 0; t < L; ++t) {
        float kt = k[base + (size_t)t * C];
        float vt = v[base + (size_t)t * C];
        float ww = u + kt;
        float q  = fmaxf(pp, ww);
        float e1 = __expf(pp - q);
        float e2 = __expf(ww - q);
        wkv[base + (size_t)t * C] = (e1 * aa + e2 * vt) / (e1 * bb + e2);
        float ww2 = pp + w;
        float q2  = fmaxf(ww2, kt);
        e1 = __expf(ww2 - q2);
        e2 = __expf(kt - q2);
        aa = e1 * aa + e2 * vt;
        bb = e1 * bb + e2;
        pp = q2;
    }
}

// ---------------------------------------------------------------------------
// 5) rwkv = sigmoid(r) * wkv -> bf16
// ---------------------------------------------------------------------------
__global__ void rwkv_prep(const float* __restrict__ r, const float* __restrict__ wkv,
                          unsigned short* __restrict__ out, size_t n4) {
    size_t i = (size_t)blockIdx.x * blockDim.x + threadIdx.x;
    if (i >= n4) return;
    float4 rv = ((const float4*)r)[i];
    float4 wv = ((const float4*)wkv)[i];
    ushort4 o;
    o.x = f2bfu(wv.x / (1.f + __expf(-rv.x)));
    o.y = f2bfu(wv.y / (1.f + __expf(-rv.y)));
    o.z = f2bfu(wv.z / (1.f + __expf(-rv.z)));
    o.w = f2bfu(wv.w / (1.f + __expf(-rv.w)));
    ((ushort4*)out)[i] = o;
}

// ---------------------------------------------------------------------------
extern "C" void kernel_launch(void* const* d_in, const int* in_sizes, int n_in,
                              void* d_out, int out_size, void* d_ws, size_t ws_size,
                              hipStream_t stream) {
    const int Bn = 8, T = 2048, C = 1024, S = 16;
    const int M = Bn * T, K = C;

    const float* x    = (const float*)d_in[0];
    const float* tdec = (const float*)d_in[1];
    const float* tfir = (const float*)d_in[2];
    const float* tmk  = (const float*)d_in[3];
    const float* tmv  = (const float*)d_in[4];
    const float* tmr  = (const float*)d_in[5];
    const float* Wk   = (const float*)d_in[6];
    const float* Wv   = (const float*)d_in[7];
    const float* Wr   = (const float*)d_in[8];
    const float* Wo   = (const float*)d_in[9];

    char* ws = (char*)d_ws;
    const size_t MB = 1024ull * 1024ull;
    unsigned short* wsWk = (unsigned short*)(ws + 0 * MB);     // 2MB each
    unsigned short* wsWv = (unsigned short*)(ws + 2 * MB);
    unsigned short* wsWr = (unsigned short*)(ws + 4 * MB);
    unsigned short* wsWo = (unsigned short*)(ws + 6 * MB);
    unsigned short* xk   = (unsigned short*)(ws + 8 * MB);     // 32MB each
    unsigned short* xv   = (unsigned short*)(ws + 40 * MB);
    unsigned short* xr   = (unsigned short*)(ws + 72 * MB);
    float* kb            = (float*)(ws + 104 * MB);            // 64MB each
    float* vb            = (float*)(ws + 168 * MB);
    float* rb            = (float*)(ws + 232 * MB);
    float* wkv           = (float*)(ws + 8 * MB);              // aliases xk+xv (dead)
    unsigned short* rwkvb = (unsigned short*)(ws + 72 * MB);   // aliases xr (dead)
    float* stA           = (float*)(ws + 296 * MB);
    float* stB = stA + (size_t)S * Bn * C;
    float* stP = stB + (size_t)S * Bn * C;
    float* inA = stP + (size_t)S * Bn * C;
    float* inB = inA + (size_t)S * Bn * C;
    float* inP = inB + (size_t)S * Bn * C;

    // 1) weight swizzle (cheap: 0.5M dwords per matrix)
    int wdw = C * K / 2;
    weight_swizzle<<<(wdw + 255) / 256, 256, 0, stream>>>(Wk, wsWk, K, wdw);
    weight_swizzle<<<(wdw + 255) / 256, 256, 0, stream>>>(Wv, wsWv, K, wdw);
    weight_swizzle<<<(wdw + 255) / 256, 256, 0, stream>>>(Wr, wsWr, K, wdw);
    weight_swizzle<<<(wdw + 255) / 256, 256, 0, stream>>>(Wo, wsWo, K, wdw);

    // 2) time-shift + mix
    size_t n4 = (size_t)M * C / 4;
    time_mix_prep<<<(unsigned)((n4 + 255) / 256), 256, 0, stream>>>(
        x, tmk, tmv, tmr, xk, xv, xr, T, C, n4);

    // 3) projections (bf16 WMMA)
    dim3 ggrid(GEMM_N / NBLK, GEMM_M / MBLK);
    gemm_bf16_wmma<<<ggrid, 256, 0, stream>>>(xk, wsWk, kb);
    gemm_bf16_wmma<<<ggrid, 256, 0, stream>>>(xv, wsWv, vb);
    gemm_bf16_wmma<<<ggrid, 256, 0, stream>>>(xr, wsWr, rb);

    // 4) chunked WKV scan
    wkv_pass1<<<(S * Bn * C) / 256, 256, 0, stream>>>(kb, vb, tdec, stA, stB, stP, Bn, T, C, S);
    wkv_pass2<<<(Bn * C) / 256, 256, 0, stream>>>(stA, stB, stP, inA, inB, inP, tdec, Bn, T, C, S);
    wkv_pass3<<<(S * Bn * C) / 256, 256, 0, stream>>>(kb, vb, inA, inB, inP, tdec, tfir, wkv, Bn, T, C, S);

    // 5) gate
    rwkv_prep<<<(unsigned)((n4 + 255) / 256), 256, 0, stream>>>(rb, wkv, rwkvb, n4);

    // 6) output projection
    gemm_bf16_wmma<<<ggrid, 256, 0, stream>>>(rwkvb, wsWo, (float*)d_out);
}
